// SSIM_3968549782135
// MI455X (gfx1250) — matrix operand order repaired and checked by reading, not compiled, not run on hardware
//
#include <hip/hip_runtime.h>

// SSIM on MI455X (gfx1250): separable 11x11 Gaussian conv as banded matmuls
// on the f16 WMMA pipe (v_wmma_f32_16x16x32_f16, f32 accum).
// v2: coalesced b128 staging of input tiles into padded f16 LDS, gaussian
// band matrices via zero-padded LDS lookup table (no cndmask chains),
// packed-f16 products. Deterministic two-stage reduction.

typedef __attribute__((ext_vector_type(16))) _Float16 v16h;
typedef __attribute__((ext_vector_type(8)))  float    v8f;
typedef __attribute__((ext_vector_type(4)))  _Float16 v4h;
typedef __attribute__((ext_vector_type(4)))  float    v4f;

#define HW      512
#define OUTDIM  502          // 512 - 11 + 1
#define NTDIM   32           // ceil(502/16)
#define NPLANE  48           // 16 batch * 3 channels
#define NTILES  (NPLANE * NTDIM * NTDIM)
#define SP      36           // LDS stage pitch in f16 (bank stride 18 -> conflict-free)

__device__ __forceinline__ v8f wmma_f16(v16h a, v16h b, v8f c) {
  // D = A(16x32 f16) * B(32x16 f16) + C(16x16 f32)
  return __builtin_amdgcn_wmma_f32_16x16x32_f16(
      /*neg_a=*/false, a, /*neg_b=*/false, b,
      /*c_mod=*/(short)0, c, /*reuse_a=*/false, /*reuse_b=*/false);
}

__global__ void __launch_bounds__(32)
ssim_tile_kernel(const float* __restrict__ raw, const float* __restrict__ dst,
                 const float* __restrict__ win, float* __restrict__ partial)
{
  __shared__ __align__(16) _Float16 Xh[32][SP];   // staged raw tile (f16)
  __shared__ __align__(16) _Float16 Yh[32][SP];   // staged dst tile (f16)
  __shared__ __align__(32) _Float16 T[5][16][32]; // H-conv out: [map][col n][row k]
  __shared__ _Float16 gtab[64];                   // zero-padded 1D gaussian, g at [16..26]

  const int lane = threadIdx.x;          // wave32: one wave per workgroup
  const int hi   = lane >> 4;
  const int lo   = lane & 15;

  const int t     = blockIdx.x;
  const int tx    = t & 31;
  const int ty    = (t >> 5) & 31;
  const int plane = t >> 10;
  const int x0    = tx * 16;
  const int y0    = ty * 16;
  const size_t base = (size_t)plane * (HW * HW);

  // ---- gaussian lookup table: gtab[16+d] = g[d], zeros elsewhere ----
  gtab[lane]      = (_Float16)0.0f;
  gtab[32 + lane] = (_Float16)0.0f;
  if (lane < 11) {
    // win[i][j] = g[i]*g[j]  =>  g[i] = win[i][5] / sqrt(win[5][5])
    const float g5 = sqrtf(win[5 * 11 + 5]);
    gtab[16 + lane] = (_Float16)(win[lane * 11 + 5] / g5);
  }

  // ---- stage 32x32 f32 tiles -> f16 LDS, coalesced b128 loads ----
#pragma unroll
  for (int c = 0; c < 8; ++c) {
    const int flat = c * 128 + lane * 4;          // wave covers 4 rows per step
    const int row  = flat >> 5;                   // 0..31
    const int col  = flat & 31;                   // 0,4,...,28
    int grow = y0 + row; if (grow > HW - 1) grow = HW - 1;       // clamped rows/cols
    int gcol = x0 + col; if (gcol > HW - 4) gcol = HW - 4;       // only feed masked outputs
    const size_t idx = base + (size_t)grow * HW + gcol;          // 16B aligned
    const v4f xr = *(const v4f*)(raw + idx);
    const v4f yr = *(const v4f*)(dst + idx);
    v4h xh, yh;
#pragma unroll
    for (int i = 0; i < 4; ++i) { xh[i] = (_Float16)xr[i]; yh[i] = (_Float16)yr[i]; }
    *(v4h*)&Xh[row][col] = xh;   // 8B aligned (72*row + 8k)
    *(v4h*)&Yh[row][col] = yh;
  }
  __syncthreads();

  // ---- band matrices via table lookup (no compare/select chains) ----
  // Horizontal band as B operand (32x16): B[k][n] = g[k-n];
//   B layout: element j of lane L -> k = (j<8 ? j : j+8) + 8*(L/16), n = L%16.
  // Vertical band as A operand (16x32): A[m][k] = g[k-m];
//   A layout: lane m = L%16; element j -> k = j + 16*(L/16).
  v16h Bh, Av;
#pragma unroll
  for (int j = 0; j < 16; ++j) {
    const int kB = ((j < 8) ? j : (j + 8)) + 8 * hi;
    Bh[j] = gtab[16 + kB - lo];
    Av[j] = gtab[16 + j + 16 * hi - lo];
  }

  // ---- horizontal pass: two 16-row blocks -> 32 H-conv rows in T ----
#pragma unroll
  for (int rb = 0; rb < 2; ++rb) {
    const int row = 16 * rb + lo;                 // A-layout: lane holds row m = lo
    // A-layout K indices: j=0..7 -> 8*hi + j ; j=8..15 -> 16 + 8*hi + (j-8)
    const v4h xa = *(const v4h*)&Xh[row][8 * hi];
    const v4h xb = *(const v4h*)&Xh[row][8 * hi + 4];
    const v4h xc = *(const v4h*)&Xh[row][16 + 8 * hi];
    const v4h xd = *(const v4h*)&Xh[row][20 + 8 * hi];
    const v4h ya = *(const v4h*)&Yh[row][8 * hi];
    const v4h yb = *(const v4h*)&Yh[row][8 * hi + 4];
    const v4h yc = *(const v4h*)&Yh[row][16 + 8 * hi];
    const v4h yd = *(const v4h*)&Yh[row][20 + 8 * hi];
    v16h ax, ay;
#pragma unroll
    for (int i = 0; i < 4; ++i) {
      ax[i] = xa[i]; ax[4 + i] = xb[i]; ax[8 + i] = xc[i]; ax[12 + i] = xd[i];
      ay[i] = ya[i]; ay[4 + i] = yb[i]; ay[8 + i] = yc[i]; ay[12 + i] = yd[i];
    }
    const v16h axx = ax * ax;     // packed f16 products
    const v16h ayy = ay * ay;
    const v16h axy = ax * ay;
    const v16h amat[5] = { ax, ay, axx, ayy, axy };
#pragma unroll
    for (int m = 0; m < 5; ++m) {
      v8f c{};
      c = wmma_f16(amat[m], Bh, c);
      // C layout: c[r] is (row = r + 8*hi, col = lo) within this block.
#pragma unroll
      for (int r = 0; r < 8; ++r)
        T[m][lo][16 * rb + r + 8 * hi] = (_Float16)c[r];
    }
  }

  __syncthreads();

  // ---- vertical pass: B operand = H-conv slab from T ----
  v8f res[5];
#pragma unroll
  for (int m = 0; m < 5; ++m) {
    // element j needs T[m][lo][j + 16*hi]: 16 contiguous f16, 32B aligned
    const v16h b = *(const v16h*)&T[m][lo][16 * hi];
    v8f c{};
    res[m] = wmma_f16(Av, b, c);
  }

  // ---- elementwise SSIM map + masked accumulation ----
  const float C1 = 6.5025f;     // (0.01*255)^2
  const float C2 = 58.5225f;    // (0.03*255)^2
  float acc = 0.0f;
#pragma unroll
  for (int r = 0; r < 8; ++r) {
    const float mu1 = res[0][r], mu2 = res[1][r];
    const float exx = res[2][r], eyy = res[3][r], exy = res[4][r];
    const float mu1s = mu1 * mu1, mu2s = mu2 * mu2, mu12 = mu1 * mu2;
    const float v1 = exx - mu1s, v2 = eyy - mu2s, cv = exy - mu12;
    const float num = (2.0f * mu12 + C1) * (2.0f * cv + C2);
    const float den = (mu1s + mu2s + C1) * (v1 + v2 + C2);
    const int yy = y0 + r + 8 * hi;
    const int xx = x0 + lo;
    if (yy < OUTDIM && xx < OUTDIM) acc += num / den;
  }

  // wave32 reduction
#pragma unroll
  for (int off = 16; off > 0; off >>= 1) acc += __shfl_down(acc, off);
  if (lane == 0)
    partial[blockIdx.x] = acc * (1.0f / (48.0f * 502.0f * 502.0f));
}

// Deterministic final reduction: fixed traversal order, single block.
__global__ void __launch_bounds__(256)
ssim_reduce_kernel(const float* __restrict__ in, float* __restrict__ out, int n)
{
  __shared__ float s[256];
  float a = 0.0f;
  for (int i = (int)threadIdx.x; i < n; i += 256) a += in[i];
  s[threadIdx.x] = a;
  __syncthreads();
  for (int off = 128; off > 0; off >>= 1) {
    if ((int)threadIdx.x < off) s[threadIdx.x] += s[threadIdx.x + off];
    __syncthreads();
  }
  if (threadIdx.x == 0) out[0] = s[0];
}

extern "C" void kernel_launch(void* const* d_in, const int* in_sizes, int n_in,
                              void* d_out, int out_size, void* d_ws, size_t ws_size,
                              hipStream_t stream) {
  (void)in_sizes; (void)n_in; (void)out_size; (void)ws_size;
  const float* raw = (const float*)d_in[0];   // (16,3,512,512) f32
  const float* dst = (const float*)d_in[1];   // (16,3,512,512) f32
  const float* win = (const float*)d_in[2];   // (11,11) f32
  float* partial = (float*)d_ws;              // NTILES floats (192 KB)

  ssim_tile_kernel<<<NTILES, 32, 0, stream>>>(raw, dst, win, partial);
  ssim_reduce_kernel<<<1, 256, 0, stream>>>(partial, (float*)d_out, NTILES);
}
